// GlobalContextAttention_58720792871193
// MI455X (gfx1250) — matrix operand order repaired and checked by reading, not compiled, not run on hardware
//
#include <hip/hip_runtime.h>

typedef __bf16 bf16_t;
typedef __attribute__((ext_vector_type(16))) __bf16 v16bf;
typedef __attribute__((ext_vector_type(8)))  __bf16 v8bf;
typedef __attribute__((ext_vector_type(4)))  __bf16 v4bf;
typedef __attribute__((ext_vector_type(8)))  float  v8f;
typedef __attribute__((ext_vector_type(4)))  float  v4f;

#define SEQ    2048
#define DMODEL 1024
#define DK     64
#define NHEAD  16

// ---------- helpers ----------

__device__ __forceinline__ bf16_t cvtb(float f) {
  return static_cast<bf16_t>(f);   // let clang use native gfx1250 bf16 cvt
}

__device__ __forceinline__ v8f wmma_bf16(v16bf a, v16bf b, v8f c) {
  return __builtin_amdgcn_wmma_f32_16x16x32_bf16(
      false, a, false, b, (short)0, c, false, false);
}

// A-fragment (16x32 bf16, M x K): lane holds its row (caller applies via p),
// K elements { hi*8 .. +7 } and { 16+hi*8 .. +7 }.
__device__ __forceinline__ v16bf load_afrag_bf16(const bf16_t* p, int hi) {
  v16bf a;
  v8bf x = *(const v8bf*)(p + hi * 8);
  v8bf y = *(const v8bf*)(p + 16 + hi * 8);
#pragma unroll
  for (int i = 0; i < 8; ++i) { a[i] = x[i]; a[8 + i] = y[i]; }
  return a;
}

// B-fragment (32x16 bf16, K x N): lane holds column n (caller applies via p),
// K elements { hi*8 .. +7 } and { 16+hi*8 .. +7 }.
__device__ __forceinline__ v16bf load_bfrag_bf16(const bf16_t* p, int hi) {
  v16bf b;
  v8bf x = *(const v8bf*)(p + hi * 8);
  v8bf y = *(const v8bf*)(p + 16 + hi * 8);
#pragma unroll
  for (int i = 0; i < 8; ++i) { b[i] = x[i]; b[8 + i] = y[i]; }
  return b;
}

// ---------- elementwise f32 -> bf16 converter (runs once per matrix) ----------
__global__ void __launch_bounds__(256)
cvt_f32_bf16(const float* __restrict__ in, bf16_t* __restrict__ out, int n4) {
  int i = blockIdx.x * blockDim.x + threadIdx.x;
  if (i < n4) {
    v4f x = ((const v4f*)in)[i];
    v4bf y;
#pragma unroll
    for (int j = 0; j < 4; ++j) y[j] = cvtb(x[j]);
    ((v4bf*)out)[i] = y;
  }
}

// ---------- GEMM: Y[M,N] = A[M,K](bf16) @ W[N,K](bf16)^T + bias ----------
// mode 0: store bf16 into [b,h,n,dk]          (Q / K projections)
// mode 1: store bf16 into [b,h,dk,n], * vw    (V projection, transposed+weighted)
// mode 2: store f32 row-major [M,N]           (output projection -> d_out)
__global__ void __launch_bounds__(256)
gemm_bf16(const bf16_t* __restrict__ A, const bf16_t* __restrict__ W,
          const float* __restrict__ bias, const float* __restrict__ vw,
          void* __restrict__ outp, int mode, int M, int N, int K)
{
  const int lane = threadIdx.x & 31;
  const int wave = threadIdx.x >> 5;
  const int ln = lane & 15, hi = lane >> 4;
  const int wm = wave >> 1, wn = wave & 1;
  const int m0 = blockIdx.y * 128 + wm * 32;
  const int n0 = blockIdx.x * 128 + wn * 64;

  const bf16_t* Arow = A + (size_t)(m0 + ln) * K;
  const bf16_t* Wcol = W + (size_t)(n0 + ln) * K;

  v8f acc[2][4] = {};

#pragma unroll 2
  for (int kk = 0; kk < K; kk += 32) {
    // speculative prefetch of next K tile (global_prefetch_b8)
    if (kk + 32 < K) {
      __builtin_prefetch(Arow + kk + 32, 0, 1);
      __builtin_prefetch(Wcol + kk + 32, 0, 1);
    }
    v16bf afrag[2];
#pragma unroll
    for (int mi = 0; mi < 2; ++mi)
      afrag[mi] = load_afrag_bf16(Arow + (size_t)mi * 16 * K + kk, hi);
    v16bf bfrag[4];
#pragma unroll
    for (int ni = 0; ni < 4; ++ni)
      bfrag[ni] = load_bfrag_bf16(Wcol + (size_t)ni * 16 * K + kk, hi);
#pragma unroll
    for (int mi = 0; mi < 2; ++mi)
#pragma unroll
      for (int ni = 0; ni < 4; ++ni)
        acc[mi][ni] = wmma_bf16(afrag[mi], bfrag[ni], acc[mi][ni]);
  }

#pragma unroll
  for (int mi = 0; mi < 2; ++mi) {
#pragma unroll
    for (int ni = 0; ni < 4; ++ni) {
#pragma unroll
      for (int v = 0; v < 8; ++v) {
        const int row = m0 + mi * 16 + v + 8 * hi;
        const int col = n0 + ni * 16 + ln;
        float val = acc[mi][ni][v] + bias[col];
        if (mode == 2) {
          ((float*)outp)[(size_t)row * N + col] = val;
        } else {
          const int b = row >> 11, nr = row & (SEQ - 1);
          const int h = col >> 6,  d  = col & (DK - 1);
          if (mode == 1) {
            val *= vw[row];
            ((bf16_t*)outp)[(((size_t)b * NHEAD + h) * DK + d) * SEQ + nr] = cvtb(val);
          } else {
            ((bf16_t*)outp)[(((size_t)b * NHEAD + h) * SEQ + nr) * DK + d] = cvtb(val);
          }
        }
      }
    }
  }
}

// ---------- Flash attention: per (bh, 64-row q tile) block, 4 waves ----------
__global__ void __launch_bounds__(128)
attn_flash(const bf16_t* __restrict__ Qb, const bf16_t* __restrict__ Kb,
           const bf16_t* __restrict__ Vt, const unsigned char* __restrict__ mask,
           bf16_t* __restrict__ ctx)
{
  __shared__ bf16_t Psh[4][16 * 32];   // per-wave P tile (16 q x 32 keys)

  const int lane = threadIdx.x & 31;
  const int wave = threadIdx.x >> 5;
  const int ln = lane & 15, hi = lane >> 4;
  const int bh = blockIdx.y;
  const int b = bh >> 4, h = bh & 15;
  const int q0 = blockIdx.x * 64 + wave * 16;

  const bf16_t* Qrow = Qb + ((size_t)bh * SEQ + q0 + ln) * DK;
  v16bf qa[2];
  qa[0] = load_afrag_bf16(Qrow + 0, hi);
  qa[1] = load_afrag_bf16(Qrow + 32, hi);

  v8f acc[4] = {};
  float mrow[8], lrow[8];
#pragma unroll
  for (int v = 0; v < 8; ++v) { mrow[v] = -3.0e38f; lrow[v] = 0.f; }

  const float scale = 0.125f;  // 1/sqrt(64)

  for (int k0 = 0; k0 < SEQ; k0 += 32) {
    const bf16_t* Kbase = Kb + ((size_t)bh * SEQ + k0) * DK;
    v8f s[2] = {};
#pragma unroll
    for (int kt = 0; kt < 2; ++kt) {
      const bf16_t* Krow = Kbase + (size_t)(kt * 16 + ln) * DK;
#pragma unroll
      for (int c = 0; c < 2; ++c) {
        v16bf kbf = load_bfrag_bf16(Krow + c * 32, hi);
        s[kt] = wmma_bf16(qa[c], kbf, s[kt]);
      }
    }

    const bool m0v = mask[b * SEQ + k0 + ln] != 0;
    const bool m1v = mask[b * SEQ + k0 + 16 + ln] != 0;

    float p0[8], p1[8];
#pragma unroll
    for (int v = 0; v < 8; ++v) {
      float a0 = m0v ? s[0][v] * scale : -1.0e9f;
      float a1 = m1v ? s[1][v] * scale : -1.0e9f;
      float t = fmaxf(a0, a1);
#pragma unroll
      for (int m = 1; m < 16; m <<= 1) t = fmaxf(t, __shfl_xor(t, m, 16));
      const float mnew  = fmaxf(mrow[v], t);
      const float alpha = __expf(mrow[v] - mnew);
      const float e0 = __expf(a0 - mnew);
      const float e1 = __expf(a1 - mnew);
      float rs = e0 + e1;
#pragma unroll
      for (int m = 1; m < 16; m <<= 1) rs += __shfl_xor(rs, m, 16);
      lrow[v] = lrow[v] * alpha + rs;
      mrow[v] = mnew;
#pragma unroll
      for (int t4 = 0; t4 < 4; ++t4) acc[t4][v] *= alpha;
      p0[v] = e0; p1[v] = e1;
    }

    // D-layout -> A-layout via LDS (same-wave round trip; DS ops are in-order)
#pragma unroll
    for (int v = 0; v < 8; ++v) {
      const int r = v + 8 * hi;
      Psh[wave][r * 32 + ln]      = cvtb(p0[v]);
      Psh[wave][r * 32 + 16 + ln] = cvtb(p1[v]);
    }
    v16bf pa;
    {
      const int o0 = ln * 32 + hi * 8;
      const int o1 = ln * 32 + 16 + hi * 8;
#pragma unroll
      for (int i = 0; i < 8; ++i) {
        pa[i]     = Psh[wave][o0 + i];
        pa[8 + i] = Psh[wave][o1 + i];
      }
    }

    // context: acc[t4] += P @ Vw  (Vt is [bh, dk, n]: B-frags contiguous in keys)
#pragma unroll
    for (int t4 = 0; t4 < 4; ++t4) {
      const bf16_t* Vrow = Vt + ((size_t)bh * DK + t4 * 16 + ln) * SEQ + k0;
      v16bf vbf = load_bfrag_bf16(Vrow, hi);
      acc[t4] = wmma_bf16(pa, vbf, acc[t4]);
    }
  }

#pragma unroll
  for (int v = 0; v < 8; ++v) {
    const float inv = 1.0f / lrow[v];
    const int q = q0 + v + 8 * hi;
    const size_t rowoff = ((size_t)b * SEQ + q) * DMODEL + (size_t)h * DK;
#pragma unroll
    for (int t4 = 0; t4 < 4; ++t4)
      ctx[rowoff + t4 * 16 + ln] = cvtb(acc[t4][v] * inv);
  }
}

// ---------- launch ----------
extern "C" void kernel_launch(void* const* d_in, const int* in_sizes, int n_in,
                              void* d_out, int out_size, void* d_ws, size_t ws_size,
                              hipStream_t stream) {
  const float* Q_src = (const float*)d_in[0];
  const float* K_src = (const float*)d_in[1];
  const float* V_src = (const float*)d_in[2];
  const float* vw    = (const float*)d_in[3];
  const unsigned char* mask = (const unsigned char*)d_in[4];
  const float* Wq_w = (const float*)d_in[5];
  const float* Wq_b = (const float*)d_in[6];
  const float* Wk_w = (const float*)d_in[7];
  const float* Wk_b = (const float*)d_in[8];
  const float* Wv_w = (const float*)d_in[9];
  const float* Wv_b = (const float*)d_in[10];
  const float* Ow_w = (const float*)d_in[11];
  const float* Ow_b = (const float*)d_in[12];
  float* out = (float*)d_out;

  const int M = 2 * SEQ, N = DMODEL, K = DMODEL;
  const int nX = M * K;          // 4M activation elements
  const int nW = N * K;          // 1M weight elements

  char* ws = (char*)d_ws;
  const size_t seg = (size_t)M * DMODEL * sizeof(bf16_t);   // 8 MB
  bf16_t* Qb  = (bf16_t*)(ws + 0 * seg);           // [b,h,n,dk]
  bf16_t* Kb  = (bf16_t*)(ws + 1 * seg);           // [b,h,n,dk]
  bf16_t* Vt  = (bf16_t*)(ws + 2 * seg);           // [b,h,dk,n], value-weighted
  bf16_t* Cx  = (bf16_t*)(ws + 3 * seg);           // [b*n, d_model]
  bf16_t* Xbf = (bf16_t*)(ws + 4 * seg);           // bf16 activations (reused)
  bf16_t* Wbf = (bf16_t*)(ws + 5 * seg);           // bf16 weights     (reused)

  dim3 ggrid(N / 128, M / 128);                    // (8, 32)
  const int cbX = (nX / 4 + 255) / 256;
  const int cbW = (nW / 4 + 255) / 256;

  // Q projection
  cvt_f32_bf16<<<cbX, 256, 0, stream>>>(Q_src, Xbf, nX / 4);
  cvt_f32_bf16<<<cbW, 256, 0, stream>>>(Wq_w, Wbf, nW / 4);
  gemm_bf16<<<ggrid, 256, 0, stream>>>(Xbf, Wbf, Wq_b, nullptr, Qb, 0, M, N, K);
  // K projection
  cvt_f32_bf16<<<cbX, 256, 0, stream>>>(K_src, Xbf, nX / 4);
  cvt_f32_bf16<<<cbW, 256, 0, stream>>>(Wk_w, Wbf, nW / 4);
  gemm_bf16<<<ggrid, 256, 0, stream>>>(Xbf, Wbf, Wk_b, nullptr, Kb, 0, M, N, K);
  // V projection (transposed + value-weighted store)
  cvt_f32_bf16<<<cbX, 256, 0, stream>>>(V_src, Xbf, nX / 4);
  cvt_f32_bf16<<<cbW, 256, 0, stream>>>(Wv_w, Wbf, nW / 4);
  gemm_bf16<<<ggrid, 256, 0, stream>>>(Xbf, Wbf, Wv_b, vw, Vt, 1, M, N, K);

  attn_flash<<<dim3(SEQ / 64, 2 * NHEAD), 128, 0, stream>>>(Qb, Kb, Vt, mask, Cx);

  // output projection -> f32 d_out
  cvt_f32_bf16<<<cbW, 256, 0, stream>>>(Ow_w, Wbf, nW / 4);
  gemm_bf16<<<ggrid, 256, 0, stream>>>(Cx, Wbf, Ow_b, nullptr, out, 2, M, N, K);
}
